// Model_63848983822825
// MI455X (gfx1250) — matrix operand order repaired and checked by reading, not compiled, not run on hardware
//
#include <hip/hip_runtime.h>
#include <math.h>

typedef _Float16 v16h __attribute__((ext_vector_type(16)));
typedef _Float16 v8h  __attribute__((ext_vector_type(8)));
typedef float    v8f  __attribute__((ext_vector_type(8)));

#define DEV __device__ __forceinline__

DEV int Lof(int m){ return (m==0)?0:((m<4)?1:2); }
DEV int OFFof(int l){ return (l==0)?0:((l==1)?1:4); }
DEV int DIMof(int l){ return (l==0)?1:((l==1)?3:5); }

DEV v16h cat16(v8h a, v8h b){
  return __builtin_shufflevector(a,b,0,1,2,3,4,5,6,7,8,9,10,11,12,13,14,15);
}

// ---------------- constant builders (run on device, deterministic) ----------

DEV double dfact(int n){ double r=1.0; for(int i=2;i<=n;++i) r*=(double)i; return r; }

DEV double cgc(int j1,int m1,int j2,int m2,int j3,int m3){
  if(m1+m2!=m3) return 0.0;
  int lo=j1-j2; if(lo<0) lo=-lo;
  if(j3<lo || j3>j1+j2) return 0.0;
  double pre = sqrt((double)(2*j3+1)*dfact(j3+j1-j2)*dfact(j3-j1+j2)*dfact(j1+j2-j3)/dfact(j1+j2+j3+1));
  pre *= sqrt(dfact(j3+m3)*dfact(j3-m3)*dfact(j1-m1)*dfact(j1+m1)*dfact(j2-m2)*dfact(j2+m2));
  double s=0.0;
  for(int k=0;k<=j1+j2+j3;++k){
    int d0=k,d1=j1+j2-j3-k,d2=j1-m1-k,d3=j2+m2-k,d4=j3-j2+m1+k,d5=j3-j1-m2+k;
    if(d0<0||d1<0||d2<0||d3<0||d4<0||d5<0) continue;
    double den=dfact(d0)*dfact(d1)*dfact(d2)*dfact(d3)*dfact(d4)*dfact(d5);
    s += ((k&1)? -1.0:1.0)/den;
  }
  return pre*s;
}

DEV void umat(int l, double* Ur, double* Ui){
  int d=2*l+1;
  for(int i=0;i<d*d;++i){ Ur[i]=0.0; Ui[i]=0.0; }
  double s2 = 0.7071067811865476;
  for(int m=-l;m<=l;++m){
    int i=m+l;
    if(m==0){ Ur[i*d+l]=1.0; }
    else if(m>0){
      Ur[i*d+(l+m)] = ((m&1)? -1.0:1.0)*s2;
      Ur[i*d+(l-m)] = s2;
    } else {
      Ui[i*d+(l-m)] = -((((-m)&1))? -1.0:1.0)*s2;
      Ui[i*d+(l+m)] = s2;
    }
  }
}

__global__ void k_build_cg9(float* cg9){
  int t = threadIdx.x;
  if(t>=27) return;
  int lo=t/9, li=(t/3)%3, lf=t%3;
  int Do=2*lo+1, Di=2*li+1, Df=2*lf+1;
  double blk[5][5][5];
  for(int A=0;A<Di;++A)for(int Bb=0;Bb<Df;++Bb)for(int Cc=0;Cc<Do;++Cc)
    blk[A][Bb][Cc]=cgc(li,A-li,lf,Bb-lf,lo,Cc-lo);
  double Uir[25],Uii[25],Ufr[25],Ufi[25],Uor[25],Uoi[25];
  umat(li,Uir,Uii); umat(lf,Ufr,Ufi); umat(lo,Uor,Uoi);
  for(int c=0;c<Do;++c)for(int a=0;a<Di;++a)for(int b=0;b<Df;++b){
    double accr=0.0;
    for(int A=0;A<Di;++A){
      double ar=Uir[a*Di+A], ai=Uii[a*Di+A];
      for(int Bb=0;Bb<Df;++Bb){
        double br=Ufr[b*Df+Bb], bi=Ufi[b*Df+Bb];
        double pr=ar*br-ai*bi, pi=ar*bi+ai*br;
        for(int Cc=0;Cc<Do;++Cc){
          double cr=Uor[c*Do+Cc], ci=Uoi[c*Do+Cc];
          accr += blk[A][Bb][Cc]*(pr*cr+pi*ci);   // Re(p * conj(Uo))
        }
      }
    }
    cg9[(OFFof(lo)+c)*81 + (OFFof(li)+a)*9 + (OFFof(lf)+b)] = (float)accr;
  }
}

__global__ void k_build_basis(float* out, int ks){
  int vol=ks*ks*ks;
  int t = blockIdx.x*blockDim.x + threadIdx.x;
  if(t>=vol) return;
  int iz=t%ks, iy=(t/ks)%ks, ix=t/(ks*ks);
  float h=(ks-1)*0.5f;
  float X=ix-h, Y=iy-h, Z=iz-h;
  float r=sqrtf(X*X+Y*Y+Z*Z);
  float mk=(r>1e-8f)?1.f:0.f;
  float s=(r>1e-8f)?r:1.f;
  float nx=X/s, ny=Y/s, nz=Z/s;
  float Ysh[9];
  Ysh[0]=0.2820948f;
  Ysh[1]=0.4886025f*ny*mk;
  Ysh[2]=0.4886025f*nz*mk;
  Ysh[3]=0.4886025f*nx*mk;
  Ysh[4]=1.0925484f*nx*ny*mk;
  Ysh[5]=1.0925484f*ny*nz*mk;
  Ysh[6]=0.3153916f*(3.f*nz*nz-1.f)*mk;
  Ysh[7]=1.0925484f*nx*nz*mk;
  Ysh[8]=0.5462742f*(nx*nx-ny*ny)*mk;
  float step=ks*0.25f;                    // linspace(0, ks/2, 3) spacing
  float sig=fmaxf(step,0.5f);
  for(int sh=0;sh<9;++sh)
    for(int j=0;j<3;++j){
      float d=r-j*step;
      out[(sh*3+j)*vol + t] = Ysh[sh]*expf(-d*d/(2.f*sig*sig));
    }
}

__global__ void k_build_k1(const float* w1, const float* bas5, float* K1){
  int t=threadIdx.x; if(t>=81) return;
  int a=t/9, O=t%9;
  const float* wp = w1 + (a*3 + Lof(O))*3;
  for(int xyz=0;xyz<125;++xyz){
    float acc=0.f;
    for(int k=0;k<3;++k) acc += wp[k]*bas5[(O*3+k)*125+xyz];
    K1[t*125+xyz]=acc;
  }
}

// type-2 kernel -> f16 weight matrix Wk[Npad][Kpad], N = Co*9, K = Ci*9*ks^3
__global__ void k_build_wk(const float* w, const float* cg9, const float* bas, _Float16* wk,
                           int Ci, int ks, int N, int K, long Kpad, long total){
  long t=(long)blockIdx.x*blockDim.x+threadIdx.x;
  if(t>=total) return;
  long n=t/Kpad; int r=(int)(t%Kpad);
  float val=0.f;
  if(n<N && r<K){
    int vol=ks*ks*ks;
    int a=(int)(n/9), O=(int)(n%9);
    int i=r/(9*vol); int rem=r%(9*vol); int I=rem/vol; int xyz=rem%vol;
    int lo=Lof(O), li=Lof(I);
    for(int F=0;F<9;++F){
      float cg=cg9[O*81+I*9+F];
      if(cg!=0.f){
        int lf=Lof(F);
        const float* wp = w + ((((long)(a*Ci+i)*3+lo)*3+li)*3+lf)*3;
        const float* bp = bas + (long)(F*3)*vol + xyz;
        val += cg*(wp[0]*bp[0] + wp[1]*bp[vol] + wp[2]*bp[2*vol]);
      }
    }
  }
  wk[t]=(_Float16)val;
}

// ---------------- conv1 (direct, tiny) --------------------------------------

__global__ void k_conv1(const float* x, const float* K1, float* h1, long total){
  long t=(long)blockIdx.x*blockDim.x+threadIdx.x;
  if(t>=total) return;
  int s=(int)(t%32768); long tmp=t/32768; int oc=(int)(tmp%81); int b=(int)(tmp/81);
  int oz=s%32, oy=(s/32)%32, ox=s/1024;
  const float* xb=x+(long)b*32768;
  const float* kk=K1+oc*125;
  float acc=0.f;
  for(int dx=0;dx<5;++dx){ int ix=ox-2+dx; if(ix<0||ix>=32) continue;
    for(int dy=0;dy<5;++dy){ int iy=oy-2+dy; if(iy<0||iy>=32) continue;
      for(int dz=0;dz<5;++dz){ int iz=oz-2+dz; if(iz<0||iz>=32) continue;
        acc += kk[(dx*5+dy)*5+dz]*xb[(ix*32+iy)*32+iz];
      }}}
  h1[t]=acc;
}

// ---------------- CG bilinear nonlinearity ----------------------------------

__global__ void k_cg_y(const float* cg9, const float* x, float* y, int S, long total){
  long t=(long)blockIdx.x*blockDim.x+threadIdx.x;
  if(t>=total) return;
  int s=(int)(t%S); long bc=t/S;
  const float* xb=x+bc*9*(long)S+s;
  float xi[9];
  for(int i=0;i<9;++i) xi[i]=xb[(long)i*S];
  float* yb=y+bc*9*(long)S+s;
  for(int o=0;o<9;++o){
    const float* cgo=cg9+o*81;
    float acc=0.f;
    for(int i=0;i<9;++i){ float q=xi[i];
      for(int f=0;f<9;++f) acc += cgo[i*9+f]*q*xi[f]; }
    yb[(long)o*S]=acc;
  }
}

__global__ void k_cg_mix(const float* wmix, const float* y, float* out, int C, int S, long total){
  long t=(long)blockIdx.x*blockDim.x+threadIdx.x;
  if(t>=total) return;
  int s=(int)(t%S); long tmp=t/S; int m=(int)(tmp%9); tmp/=9; int d=(int)(tmp%C); int b=(int)(tmp/C);
  int l=Lof(m);
  const float* wl=wmix+((long)l*C+d)*C;
  float acc=0.f;
  for(int c=0;c<C;++c) acc += wl[c]*y[((long)(b*C+c)*9+m)*S+s];
  out[((long)(b*C+d)*9+m)*S+s]=acc;
}

// ---------------- implicit GEMM path ----------------------------------------

__global__ void k_im2col(const float* in, _Float16* A, int C9, int Din, int ks, int stride,
                         int pad, int Dout, int K, long Kpad, long total){
  long t=(long)blockIdx.x*blockDim.x+threadIdx.x;
  if(t>=total) return;
  long p=t/Kpad; int r=(int)(t%Kpad);
  float val=0.f;
  if(r<K){
    int vol=ks*ks*ks;
    int Dv=Dout*Dout*Dout;
    int b=(int)(p/Dv); int s=(int)(p%Dv);
    int oz=s%Dout, oy=(s/Dout)%Dout, ox=s/(Dout*Dout);
    int c=r/vol; int tt=r%vol;
    int dz=tt%ks, dy=(tt/ks)%ks, dx=tt/(ks*ks);
    int ix=ox*stride-pad+dx, iy=oy*stride-pad+dy, iz=oz*stride-pad+dz;
    if(ix>=0&&ix<Din&&iy>=0&&iy<Din&&iz>=0&&iz<Din)
      val=in[(((long)(b*C9+c)*Din+ix)*Din+iy)*Din+iz];
  }
  A[t]=(_Float16)val;
}

// C[M x Npad] = A[M x Kpad] * B[Npad x Kpad]^T  (f16 in, f32 out)
// block = 256 threads (8 waves): 128 (M) x 64 (N) block tile.
// wave = 32x32 macro-tile (2x2 WMMA tiles). B panel staged to LDS with
// global_load_async_to_lds_b128 (ASYNCcnt) in 256-half K chunks.
#define GEMM_CH  256
#define GEMM_BST 264   // LDS row pitch in halves (bank-spread, 16B aligned)
__global__ void __launch_bounds__(256) k_gemm_f16(const _Float16* __restrict__ A,
                                                  const _Float16* __restrict__ Bm,
                                                  float* __restrict__ Cm,
                                                  long Kpad, int Npad){
  __shared__ __align__(16) _Float16 Bs[64*GEMM_BST];
  int tid=threadIdx.x;
  int wave=tid>>5, lane=tid&31;
  int hi=lane>>4, ml=lane&15;
  int waveM=wave>>1, waveN=wave&1;
  long row0=(long)blockIdx.y*128 + waveM*32;
  int colg=blockIdx.x*64;
  const _Float16* Ar0=A + (row0+ml)*Kpad;
  const _Float16* Ar1=A + (row0+16+ml)*Kpad;
  v8f acc00,acc01,acc10,acc11;
  for(int i=0;i<8;++i){acc00[i]=0.f;acc01[i]=0.f;acc10[i]=0.f;acc11[i]=0.f;}
  const _Float16* bs0=&Bs[(waveN*32+ml)*GEMM_BST + hi*16];
  const _Float16* bs1=&Bs[(waveN*32+16+ml)*GEMM_BST + hi*16];
  for(long k0=0;k0<Kpad;k0+=GEMM_CH){
    int kc=(int)((Kpad-k0)<GEMM_CH ? (Kpad-k0) : GEMM_CH);
    __syncthreads();                       // prior chunk fully consumed
    // stage B chunk (64 cols x kc halves) into LDS asynchronously
    for(int u=0;u<8;++u){
      int hidx=(tid+u*256)*8;
      int n=hidx>>8;
      int koff=hidx&255;
      if(koff<kc){
        unsigned lds_off=(unsigned)((n*GEMM_BST+koff)*2);
        unsigned long long gp=(unsigned long long)(size_t)(Bm + (long)(colg+n)*Kpad + k0 + koff);
        asm volatile("global_load_async_to_lds_b128 %0, %1, off"
                     :: "v"(lds_off), "v"(gp) : "memory");
      }
    }
    asm volatile("s_wait_asynccnt 0x0" ::: "memory");
    __syncthreads();                       // whole panel visible to all waves
    for(int kk=0;kk<kc;kk+=32){
      long ka=k0+kk;
      __builtin_prefetch(Ar0+ka+GEMM_CH,0,1);   // global_prefetch_b8
      __builtin_prefetch(Ar1+ka+GEMM_CH,0,1);
      v16h avA=cat16(*(const v8h*)(Ar0+ka+hi*8), *(const v8h*)(Ar0+ka+16+hi*8));
      v16h avB=cat16(*(const v8h*)(Ar1+ka+hi*8), *(const v8h*)(Ar1+ka+16+hi*8));
      v16h bv0=cat16(*(const v8h*)(bs0+kk-hi*8), *(const v8h*)(bs0+kk+16-hi*8));
      v16h bv1=cat16(*(const v8h*)(bs1+kk-hi*8), *(const v8h*)(bs1+kk+16-hi*8));
      acc00=__builtin_amdgcn_wmma_f32_16x16x32_f16(false,avA,false,bv0,(short)0,acc00,false,false);
      acc01=__builtin_amdgcn_wmma_f32_16x16x32_f16(false,avA,false,bv1,(short)0,acc01,false,false);
      acc10=__builtin_amdgcn_wmma_f32_16x16x32_f16(false,avB,false,bv0,(short)0,acc10,false,false);
      acc11=__builtin_amdgcn_wmma_f32_16x16x32_f16(false,avB,false,bv1,(short)0,acc11,false,false);
    }
  }
  int cb=colg+waveN*32;
  {
    float* p=Cm + (row0+hi*8)*Npad + cb + ml;
    for(int v=0;v<8;++v) p[(long)v*Npad]=acc00[v];
  }
  {
    float* p=Cm + (row0+hi*8)*Npad + cb + 16 + ml;
    for(int v=0;v<8;++v) p[(long)v*Npad]=acc01[v];
  }
  {
    float* p=Cm + (row0+16+hi*8)*Npad + cb + ml;
    for(int v=0;v<8;++v) p[(long)v*Npad]=acc10[v];
  }
  {
    float* p=Cm + (row0+16+hi*8)*Npad + cb + 16 + ml;
    for(int v=0;v<8;++v) p[(long)v*Npad]=acc11[v];
  }
}

// (B*S, Npad) -> (B, N, S)
__global__ void k_untile(const float* Cm, float* out, int N, int Npad, int S, long total){
  long t=(long)blockIdx.x*blockDim.x+threadIdx.x;
  if(t>=total) return;
  int s=(int)(t%S); long tmp=t/S; int n=(int)(tmp%N); int b=(int)(tmp/N);
  out[t]=Cm[((long)(b*S+s))*Npad+n];
}

// ---------------- norms -----------------------------------------------------

__global__ void k_sbn_reduce(const float* x, float* nrm, int C, int S){
  int c=blockIdx.x/3, l=blockIdx.x%3;
  int dim=DIMof(l), off=OFFof(l);
  long total=(long)8*dim*S;
  float acc=0.f;
  for(long i=threadIdx.x;i<total;i+=blockDim.x){
    int s=(int)(i%S); long tmp=i/S; int mm=(int)(tmp%dim); int b=(int)(tmp/dim);
    float v=x[((long)(b*C+c)*9+off+mm)*S+s];
    acc+=v*v;
  }
  __shared__ float red[256];
  red[threadIdx.x]=acc; __syncthreads();
  for(int st=128;st>0;st>>=1){ if(threadIdx.x<st) red[threadIdx.x]+=red[threadIdx.x+st]; __syncthreads(); }
  if(threadIdx.x==0) nrm[blockIdx.x]=sqrtf(red[0]/(float)total+1e-5f);
}

__global__ void k_sbn_apply(float* x, const float* nrm, int C, int S, long total){
  long t=(long)blockIdx.x*blockDim.x+threadIdx.x;
  if(t>=total) return;
  long tmp=t/S; int m=(int)(tmp%9); int c=(int)((tmp/9)%C);
  x[t]/=nrm[c*3+Lof(m)];
}

// ---------------- equivariant encoder ---------------------------------------

__global__ void k_deg_linear(const float* W, const float* x, float* out, int C, int S, long total){
  long t=(long)blockIdx.x*blockDim.x+threadIdx.x;
  if(t>=total) return;
  int s=(int)(t%S); long tmp=t/S; int m=(int)(tmp%9); tmp/=9; int d=(int)(tmp%C); int b=(int)(tmp/C);
  const float* Wl=W+((long)Lof(m)*C+d)*C;
  float acc=0.f;
  for(int c=0;c<C;++c) acc+=Wl[c]*x[((long)(b*C+c)*9+m)*S+s];
  out[t]=acc;
}

__global__ void k_scores(const float* q, const float* kx, float* sc, int C, int H, int hd,
                         int S, float scale, long total){
  long t=(long)blockIdx.x*blockDim.x+threadIdx.x;
  if(t>=total) return;
  int j=(int)(t%S); long tmp=t/S; int i=(int)(tmp%S); tmp/=S; int h=(int)(tmp%H); int b=(int)(tmp/H);
  float acc=0.f;
  for(int d=0;d<hd;++d){
    int c=h*hd+d;
    const float* qb=q+((long)(b*C+c)*9)*S;
    const float* kb=kx+((long)(b*C+c)*9)*S;
    for(int m=0;m<9;++m) acc+=qb[(long)m*S+i]*kb[(long)m*S+j];
  }
  sc[t]=acc*scale;
}

__global__ void k_softmax(float* sc, int S, long rows){
  long t=(long)blockIdx.x*blockDim.x+threadIdx.x;
  if(t>=rows) return;
  float* row=sc+t*(long)S;
  float mx=row[0];
  for(int j=1;j<S;++j) mx=fmaxf(mx,row[j]);
  float sum=0.f;
  for(int j=0;j<S;++j){ float e=expf(row[j]-mx); row[j]=e; sum+=e; }
  float inv=1.f/sum;
  for(int j=0;j<S;++j) row[j]*=inv;
}

__global__ void k_av(const float* sc, const float* v, float* o, int C, int H, int hd, int S, long total){
  long t=(long)blockIdx.x*blockDim.x+threadIdx.x;
  if(t>=total) return;
  int i=(int)(t%S); long tmp=t/S; int m=(int)(tmp%9); tmp/=9; int c=(int)(tmp%C); int b=(int)(tmp/C);
  int h=c/hd;
  const float* arow=sc+((long)(b*H+h)*S+i)*S;
  const float* vb=v+((long)(b*C+c)*9+m)*S;
  float acc=0.f;
  for(int j=0;j<S;++j) acc+=arow[j]*vb[j];
  o[t]=acc;
}

__global__ void k_res_eqnorm(float* h, const float* tt, int C, int S, long total){
  long t=(long)blockIdx.x*blockDim.x+threadIdx.x;
  if(t>=total) return;
  int s=(int)(t%S); int b=(int)(t/S);
  float nrm[3];
  for(int l=0;l<3;++l){
    int off=OFFof(l), dim=DIMof(l);
    float ss=0.f;
    for(int c=0;c<C;++c)
      for(int mm=0;mm<dim;++mm){
        long idx=((long)(b*C+c)*9+off+mm)*S+s;
        float v=h[idx]+tt[idx];
        ss+=v*v;
      }
    nrm[l]=sqrtf(ss/(float)(C*dim)+1e-5f);
  }
  for(int l=0;l<3;++l){
    int off=OFFof(l), dim=DIMof(l);
    float inv=1.f/nrm[l];
    for(int c=0;c<C;++c)
      for(int mm=0;mm<dim;++mm){
        long idx=((long)(b*C+c)*9+off+mm)*S+s;
        h[idx]=(h[idx]+tt[idx])*inv;
      }
  }
}

// ---------------- head ------------------------------------------------------

__global__ void k_feat(const float* C5, float* feat){
  int t=blockIdx.x*blockDim.x+threadIdx.x;
  if(t>=8*144) return;
  int co=t%144, b=t/144;
  float acc=0.f;
  for(int s=0;s<64;++s){
    const float* row=C5+((long)(b*64+s))*1344+co*9;
    float ss=0.f;
    for(int m=0;m<9;++m) ss+=row[m]*row[m];
    acc+=sqrtf(ss+1e-8f);
  }
  feat[t]=acc*(1.f/64.f);
}

__global__ void k_final(const float* feat, const float* wlin, const float* blin, float* out){
  int t=threadIdx.x; if(t>=80) return;
  int j=t%10, b=t/10;
  float acc=blin[j];
  for(int c=0;c<144;++c) acc+=feat[b*144+c]*wlin[j*144+c];
  out[t]=acc;
}

// ---------------- host orchestration ----------------------------------------

static inline unsigned gr(long n){ return (unsigned)((n+255)/256); }

static void run_encoder(float* h, float* q, float* k, float* v, float* o, float* tb, float* sc,
                        const float* Wq, const float* Wk, const float* Wv, const float* Wo,
                        int C, int H, int S, hipStream_t stream){
  int hd=C/H;
  float scale=1.f/sqrtf((float)(hd*9));
  long nlin=8L*C*9*S;
  long nsc=8L*H*S*S;
  long nrows=8L*H*S;
  long wstride=(long)3*C*C;
  for(int layer=0;layer<4;++layer){
    k_deg_linear<<<gr(nlin),256,0,stream>>>(Wq+layer*wstride,h,q,C,S,nlin);
    k_deg_linear<<<gr(nlin),256,0,stream>>>(Wk+layer*wstride,h,k,C,S,nlin);
    k_deg_linear<<<gr(nlin),256,0,stream>>>(Wv+layer*wstride,h,v,C,S,nlin);
    k_scores<<<gr(nsc),256,0,stream>>>(q,k,sc,C,H,hd,S,scale,nsc);
    k_softmax<<<gr(nrows),256,0,stream>>>(sc,S,nrows);
    k_av<<<gr(nlin),256,0,stream>>>(sc,v,o,C,H,hd,S,nlin);
    k_deg_linear<<<gr(nlin),256,0,stream>>>(Wo+layer*wstride,o,tb,C,S,nlin);
    k_res_eqnorm<<<gr(8L*S),256,0,stream>>>(h,tb,C,S,8L*S);
  }
}

extern "C" void kernel_launch(void* const* d_in, const int* in_sizes, int n_in,
                              void* d_out, int out_size, void* d_ws, size_t ws_size,
                              hipStream_t stream){
  (void)in_sizes; (void)n_in; (void)out_size; (void)ws_size;
  const float* x   =(const float*)d_in[0];
  const float* w1  =(const float*)d_in[1];
  const float* wcg1=(const float*)d_in[2];
  const float* w2  =(const float*)d_in[3];
  const float* e1q =(const float*)d_in[4];
  const float* e1k =(const float*)d_in[5];
  const float* e1v =(const float*)d_in[6];
  const float* e1o =(const float*)d_in[7];
  const float* w3  =(const float*)d_in[8];
  const float* wcg2=(const float*)d_in[9];
  const float* w4  =(const float*)d_in[10];
  const float* e2q =(const float*)d_in[11];
  const float* e2k =(const float*)d_in[12];
  const float* e2v =(const float*)d_in[13];
  const float* e2o =(const float*)d_in[14];
  const float* w5  =(const float*)d_in[15];
  const float* wlin=(const float*)d_in[16];
  const float* blin=(const float*)d_in[17];

  char* ws=(char*)d_ws;
  size_t off=0;
  auto alloc=[&](size_t bytes)->char*{ char* p=ws+off; off+=(bytes+255)&~(size_t)255; return p; };

  float*     CG9 =(float*)alloc(729*4);
  float*     BAS5=(float*)alloc(3375*4);
  float*     BAS4=(float*)alloc(1728*4);
  float*     BAS3=(float*)alloc(729*4);
  float*     BAS2=(float*)alloc(216*4);
  float*     K1  =(float*)alloc(81*125*4);
  float*     H1  =(float*)alloc((size_t)8*81*32768*4);   // conv1 out, then cg-mix out
  float*     Y1  =(float*)alloc((size_t)8*81*32768*4);
  _Float16*  WK2 =(_Float16*)alloc((size_t)192*5184*2);
  _Float16*  A2  =(_Float16*)alloc((size_t)4096*5184*2);
  float*     C2  =(float*)alloc((size_t)4096*192*4);
  float*     H2  =(float*)alloc((size_t)8*162*512*4);
  float*     Eq  =(float*)alloc((size_t)8*162*512*4);
  float*     Ek  =(float*)alloc((size_t)8*162*512*4);
  float*     Ev  =(float*)alloc((size_t)8*162*512*4);
  float*     Eo  =(float*)alloc((size_t)8*162*512*4);
  float*     Et  =(float*)alloc((size_t)8*162*512*4);
  float*     SC  =(float*)alloc((size_t)8*3*512*512*4);
  float*     NRM =(float*)alloc(256*4);
  _Float16*  WK3 =(_Float16*)alloc((size_t)384*4384*2);
  _Float16*  A3  =(_Float16*)alloc((size_t)4096*4384*2);
  float*     C3  =(float*)alloc((size_t)4096*384*4);
  float*     H3  =(float*)alloc((size_t)8*324*512*4);
  float*     Y2  =(float*)alloc((size_t)8*324*512*4);
  float*     H3b =(float*)alloc((size_t)8*324*512*4);
  _Float16*  WK4 =(_Float16*)alloc((size_t)704*2592*2);
  _Float16*  A4  =(_Float16*)alloc((size_t)512*2592*2);
  float*     C4  =(float*)alloc((size_t)512*704*4);
  float*     H4  =(float*)alloc((size_t)8*648*64*4);
  _Float16*  WK5 =(_Float16*)alloc((size_t)1344*41472*2);
  _Float16*  A5  =(_Float16*)alloc((size_t)512*41472*2);
  float*     C5  =(float*)alloc((size_t)512*1344*4);
  float*     FEAT=(float*)alloc((size_t)8*144*4);

  long n;
  // constants
  k_build_cg9<<<1,32,0,stream>>>(CG9);
  k_build_basis<<<1,128,0,stream>>>(BAS5,5);
  k_build_basis<<<1,64,0,stream>>>(BAS4,4);
  k_build_basis<<<1,32,0,stream>>>(BAS3,3);
  k_build_basis<<<1,32,0,stream>>>(BAS2,2);
  k_build_k1<<<1,128,0,stream>>>(w1,BAS5,K1);

  // conv1 (type-1, direct) -> H1 (8,81,32^3)
  n=8L*81*32768; k_conv1<<<gr(n),256,0,stream>>>(x,K1,H1,n);

  // CG nonlinearity 1 (C=9) -> H1 overwritten with mixed output
  n=8L*9*32768;   k_cg_y  <<<gr(n),256,0,stream>>>(CG9,H1,Y1,32768,n);
  n=8L*9*9*32768; k_cg_mix<<<gr(n),256,0,stream>>>(wcg1,Y1,H1,9,32768,n);

  // conv2: ks4 s4 p0, 32 -> 8  (M=4096, N=162->192, K=5184)
  n=192L*5184;  k_build_wk<<<gr(n),256,0,stream>>>(w2,CG9,BAS4,WK2,9,4,162,5184,5184,n);
  n=4096L*5184; k_im2col <<<gr(n),256,0,stream>>>(H1,A2,81,32,4,4,0,8,5184,5184,n);
  k_gemm_f16<<<dim3(3,32),256,0,stream>>>(A2,WK2,C2,5184,192);
  n=8L*162*512; k_untile<<<gr(n),256,0,stream>>>(C2,H2,162,192,512,n);

  // sbn + encoder1 (C=18, H=3, S=512)
  k_sbn_reduce<<<18*3,256,0,stream>>>(H2,NRM,18,512);
  n=8L*162*512; k_sbn_apply<<<gr(n),256,0,stream>>>(H2,NRM,18,512,n);
  run_encoder(H2,Eq,Ek,Ev,Eo,Et,SC,e1q,e1k,e1v,e1o,18,3,512,stream);

  // conv3: ks3 s1 p1, 8 -> 8  (M=4096, N=324->384, K=4374->4384)
  n=384L*4384;  k_build_wk<<<gr(n),256,0,stream>>>(w3,CG9,BAS3,WK3,18,3,324,4374,4384,n);
  n=4096L*4384; k_im2col <<<gr(n),256,0,stream>>>(H2,A3,162,8,3,1,1,8,4374,4384,n);
  k_gemm_f16<<<dim3(6,32),256,0,stream>>>(A3,WK3,C3,4384,384);
  n=8L*324*512; k_untile<<<gr(n),256,0,stream>>>(C3,H3,324,384,512,n);

  // CG nonlinearity 2 (C=36)
  n=8L*36*512;   k_cg_y  <<<gr(n),256,0,stream>>>(CG9,H3,Y2,512,n);
  n=8L*36*9*512; k_cg_mix<<<gr(n),256,0,stream>>>(wcg2,Y2,H3b,36,512,n);

  // conv4: ks2 s2 p0, 8 -> 4  (M=512, N=648->704, K=2592)
  n=704L*2592; k_build_wk<<<gr(n),256,0,stream>>>(w4,CG9,BAS2,WK4,36,2,648,2592,2592,n);
  n=512L*2592; k_im2col <<<gr(n),256,0,stream>>>(H3b,A4,324,8,2,2,0,4,2592,2592,n);
  k_gemm_f16<<<dim3(11,4),256,0,stream>>>(A4,WK4,C4,2592,704);
  n=8L*648*64; k_untile<<<gr(n),256,0,stream>>>(C4,H4,648,704,64,n);

  // sbn + encoder2 (C=72, H=9, S=64) — reuse encoder scratch
  k_sbn_reduce<<<72*3,256,0,stream>>>(H4,NRM,72,64);
  n=8L*648*64; k_sbn_apply<<<gr(n),256,0,stream>>>(H4,NRM,72,64,n);
  run_encoder(H4,Eq,Ek,Ev,Eo,Et,SC,e2q,e2k,e2v,e2o,72,9,64,stream);

  // conv5: ks4 s1 pad_lo1/pad_hi2, 4 -> 4  (M=512, N=1296->1344, K=41472)
  n=1344L*41472; k_build_wk<<<gr(n),256,0,stream>>>(w5,CG9,BAS4,WK5,72,4,1296,41472,41472,n);
  n=512L*41472;  k_im2col <<<gr(n),256,0,stream>>>(H4,A5,648,4,4,1,1,4,41472,41472,n);
  k_gemm_f16<<<dim3(21,4),256,0,stream>>>(A5,WK5,C5,41472,1344);

  // head
  k_feat<<<gr(8*144),256,0,stream>>>(C5,FEAT);
  k_final<<<1,128,0,stream>>>(FEAT,wlin,blin,(float*)d_out);
}
